// lanegcn_vanilla_gan_60155311948394
// MI455X (gfx1250) — matrix operand order repaired and checked by reading, not compiled, not run on hardware
//
#include <hip/hip_runtime.h>
#include <hip/hip_bf16.h>
#include <cstdint>

typedef __attribute__((ext_vector_type(16))) _Float16 v16h;
typedef __attribute__((ext_vector_type(8)))  float    v8f;

#define LDSTRIDE 136   // 272B row pitch -> 4-bank rotation per row, conflict-free
#define EPSGN 1e-5f

union AFrag { v16h v; uint4 u[2]; _Float16 h[16]; };

// ---------------------------------------------------------------------------
// A-fragment loaders: 16x32 f16 A layout (ISA 7.12.2):
// lane L holds row m=L&15; halves 0..7  -> k = kk*32 + (L>=16?8:0) + i
//                          halves 8..15 -> k = kk*32 + 16 + (L>=16?8:0) + (i-8)
// => two contiguous 16B chunks per lane.
// ---------------------------------------------------------------------------
__device__ __forceinline__ v16h loadA_g(const _Float16* __restrict__ row, int kk, int koff8) {
    AFrag a;
    const _Float16* p = row + kk * 32 + koff8;
    a.u[0] = *(const uint4*)(p);
    a.u[1] = *(const uint4*)(p + 16);
    return a.v;
}

__device__ __forceinline__ v16h loadA_lds(const _Float16* row, int kk, int koff8) {
    AFrag a;
    const _Float16* p = row + kk * 32 + koff8;
    a.u[0] = *(const uint4*)(p);
    a.u[1] = *(const uint4*)(p + 16);
    return a.v;
}

// ---------------------------------------------------------------------------
// One K-block pass of a 16x(32*KF) @ (32*KF)x128 GEMM, f16 in / f32 acc.
// B layout: lane L holds col n=L&15, K halves at (L>=16?8:0)+i (i<8) and
// 16+(L>=16?8:0)+(i-8) (i>=8) -> row-major [out][in] weights are already B-major.
// All 8 B fragments of a K-step are loaded as one 16-load clause (b[8] block),
// then the 8 WMMAs drain with staggered loadcnt waits instead of per-WMMA
// load->wait0->wmma serialization.
// ---------------------------------------------------------------------------
template <int KF>
__device__ __forceinline__ void gemm_pass(const v16h afr[KF], const _Float16* __restrict__ W,
                                          int Kd, int kbase, int lane, v8f acc[8]) {
    const int n0  = lane & 15;
    const int bko = (lane & 16) ? 8 : 0;
#pragma unroll
    for (int kk = 0; kk < KF; ++kk) {
        AFrag b[8];
#pragma unroll
        for (int nt = 0; nt < 8; ++nt) {
            const _Float16* bp = W + (size_t)(nt * 16 + n0) * Kd + kbase + kk * 32 + bko;
            b[nt].u[0] = *(const uint4*)(bp);
            b[nt].u[1] = *(const uint4*)(bp + 16);
        }
#pragma unroll
        for (int nt = 0; nt < 8; ++nt)
            acc[nt] = __builtin_amdgcn_wmma_f32_16x16x32_f16(
                false, afr[kk], false, b[nt].v, (short)0, acc[nt], false, false);
    }
}

// ---------------------------------------------------------------------------
// GroupNorm(1 group over 128 ch) + optional ReLU, in D-fragment registers.
// Row m lives at VGPR v=m&7 across 16 lanes of one half-wave x 8 N-tiles;
// shfl_xor masks 1,2,4,8 stay within each half-wave.
// ---------------------------------------------------------------------------
__device__ __forceinline__ void gn_rows(v8f acc[8], const float* __restrict__ g,
                                        const float* __restrict__ bt, int lane, bool do_relu) {
    const int n0 = lane & 15;
    float s1[8], s2[8];
#pragma unroll
    for (int v = 0; v < 8; ++v) { s1[v] = 0.f; s2[v] = 0.f; }
#pragma unroll
    for (int nt = 0; nt < 8; ++nt)
#pragma unroll
        for (int v = 0; v < 8; ++v) { float x = acc[nt][v]; s1[v] += x; s2[v] += x * x; }
#pragma unroll
    for (int m = 1; m < 16; m <<= 1) {
#pragma unroll
        for (int v = 0; v < 8; ++v) {
            s1[v] += __shfl_xor(s1[v], m, 32);
            s2[v] += __shfl_xor(s2[v], m, 32);
        }
    }
#pragma unroll
    for (int nt = 0; nt < 8; ++nt) {
        float gg = g[nt * 16 + n0], bb = bt[nt * 16 + n0];
#pragma unroll
        for (int v = 0; v < 8; ++v) {
            float mu  = s1[v] * (1.f / 128.f);
            float var = s2[v] * (1.f / 128.f) - mu * mu;
            float sc  = rsqrtf(var + EPSGN);
            float y   = (acc[nt][v] - mu) * sc * gg + bb;
            acc[nt][v] = do_relu ? fmaxf(y, 0.f) : y;
        }
    }
}

// D-fragment f32 -> row-major f16 tile in LDS (padded pitch)
__device__ __forceinline__ void store_lds_f16(_Float16* buf, const v8f acc[8], int lane) {
    const int n0 = lane & 15;
    const int mb = (lane & 16) ? 8 : 0;
#pragma unroll
    for (int nt = 0; nt < 8; ++nt)
#pragma unroll
        for (int v = 0; v < 8; ++v)
            buf[(mb + v) * LDSTRIDE + nt * 16 + n0] = (_Float16)acc[nt][v];
}

__device__ __forceinline__ void wait_ds() {
    asm volatile("s_wait_dscnt 0" ::: "memory");
}

// ---------------------------------------------------------------------------
// f32 -> f16 converter (weights + activations staged once into workspace)
// ---------------------------------------------------------------------------
__global__ __launch_bounds__(256) void cvt_f32_f16(const float* __restrict__ s,
                                                   _Float16* __restrict__ d, int n) {
    int i = blockIdx.x * 256 + threadIdx.x;
    if (i < n) d[i] = (_Float16)s[i];
}

// ---------------------------------------------------------------------------
// K2: out = agts @ w_a.T   (seeds d_out before edge scatter-add)
// ---------------------------------------------------------------------------
__global__ __launch_bounds__(128) void node_gemm_kernel(const _Float16* __restrict__ agts_h,
                                                        const _Float16* __restrict__ w_ah,
                                                        float* __restrict__ out, int N) {
    const int lane = threadIdx.x & 31;
    const int wid  = threadIdx.x >> 5;
    const int tile = blockIdx.x * 4 + wid;
    const int nT   = (N + 15) >> 4;
    if (tile >= nT) return;
    const int tb    = tile * 16;
    const int n0    = lane & 15;
    const int koff8 = (lane & 16) ? 8 : 0;
    int rm = tb + n0; if (rm >= N) rm = N - 1;

    v16h afr[4];
    const _Float16* arow = agts_h + (size_t)rm * 128;
#pragma unroll
    for (int kk = 0; kk < 4; ++kk) afr[kk] = loadA_g(arow, kk, koff8);

    v8f acc[8];
#pragma unroll
    for (int nt = 0; nt < 8; ++nt) acc[nt] = (v8f)(0.f);
    gemm_pass<4>(afr, w_ah, 128, 0, lane, acc);

    const int mb = (lane & 16) ? 8 : 0;
#pragma unroll
    for (int v = 0; v < 8; ++v) {
        int row = tb + mb + v;
        if (row < N)
#pragma unroll
            for (int nt = 0; nt < 8; ++nt)
                out[(size_t)row * 128 + nt * 16 + n0] = acc[nt][v];
    }
}

// ---------------------------------------------------------------------------
// K3: fused edge pipeline. One wave = 16 edges.
//   d = relu(gn(relu(diff@w_d1.T+b) @ w_d2.T))
//   q = relu(gn(agts[hi] @ w_q.T))
//   e = relu(gn([d,q,ctx[wi]] @ w_c1.T)) @ w_c2.T ;  out[hi] += e (atomics)
// ---------------------------------------------------------------------------
__global__ __launch_bounds__(128) void edge_kernel(
    const float* __restrict__ agt_ctrs, const float* __restrict__ ctx_ctrs,
    const int* __restrict__ hi, const int* __restrict__ wi,
    const float* __restrict__ w_d1, const float* __restrict__ b_d1,
    const _Float16* __restrict__ w_d2h, const float* __restrict__ g_d2, const float* __restrict__ bt_d2,
    const _Float16* __restrict__ w_qh,  const float* __restrict__ g_q,  const float* __restrict__ bt_q,
    const _Float16* __restrict__ w_c1h, const float* __restrict__ g_c1, const float* __restrict__ bt_c1,
    const _Float16* __restrict__ w_c2h,
    const _Float16* __restrict__ agts_h, const _Float16* __restrict__ ctx_h,
    float* __restrict__ out, int E) {
    __shared__ _Float16 sm[4 * 2 * 16 * LDSTRIDE];
    const int lane = threadIdx.x & 31;
    const int wid  = threadIdx.x >> 5;
    const int tile = blockIdx.x * 4 + wid;
    const int nT   = (E + 15) >> 4;
    if (tile >= nT) return;
    _Float16* dbuf = sm + wid * (2 * 16 * LDSTRIDE);
    _Float16* qbuf = dbuf + 16 * LDSTRIDE;

    const int tb    = tile * 16;
    const int n0    = lane & 15;
    const int koff8 = (lane & 16) ? 8 : 0;
    int em = tb + n0; if (em >= E) em = E - 1;
    const int hie = hi[em];
    const int wie = wi[em];
    const float dx = agt_ctrs[2 * hie]     - ctx_ctrs[2 * wie];
    const float dy = agt_ctrs[2 * hie + 1] - ctx_ctrs[2 * wie + 1];

    // ---- d1 = relu(diff @ w_d1.T + b) built directly in A layout ----
    v16h afr[4];
#pragma unroll
    for (int kk = 0; kk < 4; ++kk) {
        AFrag a;
#pragma unroll
        for (int i = 0; i < 16; ++i) {
            int k = kk * 32 + ((i >> 3) << 4) + koff8 + (i & 7);
            float v = fmaf(dx, w_d1[2 * k], fmaf(dy, w_d1[2 * k + 1], b_d1[k]));
            a.h[i] = (_Float16)fmaxf(v, 0.f);
        }
        afr[kk] = a.v;
    }

    v8f acc[8];
    // ---- d = relu(gn(d1 @ w_d2.T)) ----
#pragma unroll
    for (int nt = 0; nt < 8; ++nt) acc[nt] = (v8f)(0.f);
    gemm_pass<4>(afr, w_d2h, 128, 0, lane, acc);
    gn_rows(acc, g_d2, bt_d2, lane, true);
    store_lds_f16(dbuf, acc, lane);

    // ---- q = relu(gn(agts[hi] @ w_q.T)) ----
    const _Float16* arow = agts_h + (size_t)hie * 128;
#pragma unroll
    for (int kk = 0; kk < 4; ++kk) afr[kk] = loadA_g(arow, kk, koff8);
#pragma unroll
    for (int nt = 0; nt < 8; ++nt) acc[nt] = (v8f)(0.f);
    gemm_pass<4>(afr, w_qh, 128, 0, lane, acc);
    gn_rows(acc, g_q, bt_q, lane, true);
    store_lds_f16(qbuf, acc, lane);
    wait_ds();

    // ---- e = relu(gn([d,q,ctx[wi]] @ w_c1.T)) : 3 K-passes into one acc ----
#pragma unroll
    for (int nt = 0; nt < 8; ++nt) acc[nt] = (v8f)(0.f);
    const _Float16* drow = dbuf + n0 * LDSTRIDE;
#pragma unroll
    for (int kk = 0; kk < 4; ++kk) afr[kk] = loadA_lds(drow, kk, koff8);
    gemm_pass<4>(afr, w_c1h, 384, 0, lane, acc);
    const _Float16* qrow = qbuf + n0 * LDSTRIDE;
#pragma unroll
    for (int kk = 0; kk < 4; ++kk) afr[kk] = loadA_lds(qrow, kk, koff8);
    gemm_pass<4>(afr, w_c1h, 384, 128, lane, acc);
    const _Float16* crow = ctx_h + (size_t)wie * 128;
#pragma unroll
    for (int kk = 0; kk < 4; ++kk) afr[kk] = loadA_g(crow, kk, koff8);
    gemm_pass<4>(afr, w_c1h, 384, 256, lane, acc);
    gn_rows(acc, g_c1, bt_c1, lane, true);
    store_lds_f16(dbuf, acc, lane);   // reuse d tile for e
    wait_ds();

    // ---- e @ w_c2.T ----
#pragma unroll
    for (int kk = 0; kk < 4; ++kk) afr[kk] = loadA_lds(drow, kk, koff8);
#pragma unroll
    for (int nt = 0; nt < 8; ++nt) acc[nt] = (v8f)(0.f);
    gemm_pass<4>(afr, w_c2h, 128, 0, lane, acc);

    // ---- scatter-add into out[hi] ----
    const int mb = (lane & 16) ? 8 : 0;
#pragma unroll
    for (int v = 0; v < 8; ++v) {
        int e2 = tb + mb + v;
        if (e2 < E) {
            int row = hi[e2];
#pragma unroll
            for (int nt = 0; nt < 8; ++nt)
                atomicAdd(&out[(size_t)row * 128 + nt * 16 + n0], acc[nt][v]);
        }
    }
}

// ---------------------------------------------------------------------------
// K4: out = relu( gn( relu(gn(mid)) @ w_l.T ) + agts )   (in-place per tile)
// ---------------------------------------------------------------------------
__global__ __launch_bounds__(128) void final_kernel(const float* __restrict__ agts,
                                                    const _Float16* __restrict__ w_lh,
                                                    const float* __restrict__ g_n, const float* __restrict__ bt_n,
                                                    const float* __restrict__ g_l, const float* __restrict__ bt_l,
                                                    float* __restrict__ out, int N) {
    __shared__ _Float16 sm[4 * 16 * LDSTRIDE];
    const int lane = threadIdx.x & 31;
    const int wid  = threadIdx.x >> 5;
    const int tile = blockIdx.x * 4 + wid;
    const int nT   = (N + 15) >> 4;
    if (tile >= nT) return;
    _Float16* buf = sm + wid * (16 * LDSTRIDE);

    const int tb    = tile * 16;
    const int n0    = lane & 15;
    const int koff8 = (lane & 16) ? 8 : 0;
    const int mb    = (lane & 16) ? 8 : 0;

    v8f acc[8];
#pragma unroll
    for (int nt = 0; nt < 8; ++nt) {
#pragma unroll
        for (int v = 0; v < 8; ++v) {
            int row = tb + mb + v;
            acc[nt][v] = (row < N) ? out[(size_t)row * 128 + nt * 16 + n0] : 0.f;
        }
    }
    gn_rows(acc, g_n, bt_n, lane, true);
    store_lds_f16(buf, acc, lane);
    wait_ds();

    v16h afr[4];
    const _Float16* rrow = buf + n0 * LDSTRIDE;
#pragma unroll
    for (int kk = 0; kk < 4; ++kk) afr[kk] = loadA_lds(rrow, kk, koff8);
#pragma unroll
    for (int nt = 0; nt < 8; ++nt) acc[nt] = (v8f)(0.f);
    gemm_pass<4>(afr, w_lh, 128, 0, lane, acc);
    gn_rows(acc, g_l, bt_l, lane, false);

#pragma unroll
    for (int v = 0; v < 8; ++v) {
        int row = tb + mb + v;
        if (row < N)
#pragma unroll
            for (int nt = 0; nt < 8; ++nt) {
                float r = agts[(size_t)row * 128 + nt * 16 + n0];
                out[(size_t)row * 128 + nt * 16 + n0] = fmaxf(acc[nt][v] + r, 0.f);
            }
    }
}

// ---------------------------------------------------------------------------
extern "C" void kernel_launch(void* const* d_in, const int* in_sizes, int n_in,
                              void* d_out, int out_size, void* d_ws, size_t ws_size,
                              hipStream_t stream) {
    const float* agts     = (const float*)d_in[0];
    const float* ctx      = (const float*)d_in[1];
    const float* agt_ctrs = (const float*)d_in[2];
    const float* ctx_ctrs = (const float*)d_in[3];
    const int*   hi       = (const int*)d_in[4];
    const int*   wi       = (const int*)d_in[5];
    const float* w_d1  = (const float*)d_in[6];
    const float* b_d1  = (const float*)d_in[7];
    const float* w_d2  = (const float*)d_in[8];
    const float* g_d2  = (const float*)d_in[9];
    const float* bt_d2 = (const float*)d_in[10];
    const float* w_q   = (const float*)d_in[11];
    const float* g_q   = (const float*)d_in[12];
    const float* bt_q  = (const float*)d_in[13];
    const float* w_c1  = (const float*)d_in[14];
    const float* g_c1  = (const float*)d_in[15];
    const float* bt_c1 = (const float*)d_in[16];
    const float* w_c2  = (const float*)d_in[17];
    const float* w_a   = (const float*)d_in[18];
    const float* g_n   = (const float*)d_in[19];
    const float* bt_n  = (const float*)d_in[20];
    const float* w_l   = (const float*)d_in[21];
    const float* g_l   = (const float*)d_in[22];
    const float* bt_l  = (const float*)d_in[23];

    const int N = in_sizes[0] / 128;
    const int E = in_sizes[4];
    float* out = (float*)d_out;

    // carve f16 staging from workspace (256B aligned)
    char* wsp = (char*)d_ws;
    auto carve = [&](size_t bytes) -> _Float16* {
        _Float16* p = (_Float16*)wsp;
        wsp += (bytes + 255) & ~(size_t)255;
        return p;
    };
    _Float16* agts_h = carve((size_t)N * 128 * 2);
    _Float16* ctx_h  = carve((size_t)N * 128 * 2);
    _Float16* w_d2h  = carve(128 * 128 * 2);
    _Float16* w_qh   = carve(128 * 128 * 2);
    _Float16* w_c1h  = carve(128 * 384 * 2);
    _Float16* w_c2h  = carve(128 * 128 * 2);
    _Float16* w_ah   = carve(128 * 128 * 2);
    _Float16* w_lh   = carve(128 * 128 * 2);

    auto cvt = [&](const float* s, _Float16* d, int n) {
        cvt_f32_f16<<<(n + 255) / 256, 256, 0, stream>>>(s, d, n);
    };
    cvt(agts, agts_h, N * 128);
    cvt(ctx,  ctx_h,  N * 128);
    cvt(w_d2, w_d2h, 128 * 128);
    cvt(w_q,  w_qh,  128 * 128);
    cvt(w_c1, w_c1h, 128 * 384);
    cvt(w_c2, w_c2h, 128 * 128);
    cvt(w_a,  w_ah,  128 * 128);
    cvt(w_l,  w_lh,  128 * 128);

    const int nTn = (N + 15) / 16;
    const int nTe = (E + 15) / 16;

    node_gemm_kernel<<<(nTn + 3) / 4, 128, 0, stream>>>(agts_h, w_ah, out, N);

    edge_kernel<<<(nTe + 3) / 4, 128, 0, stream>>>(
        agt_ctrs, ctx_ctrs, hi, wi, w_d1, b_d1,
        w_d2h, g_d2, bt_d2, w_qh, g_q, bt_q,
        w_c1h, g_c1, bt_c1, w_c2h, agts_h, ctx_h, out, E);

    final_kernel<<<(nTn + 3) / 4, 128, 0, stream>>>(
        agts, w_lh, g_n, bt_n, g_l, bt_l, out, N);
}
